// CombinedModel_49125835932132
// MI455X (gfx1250) — matrix operand order repaired and checked by reading, not compiled, not run on hardware
//
#include <hip/hip_runtime.h>
#include <hip/hip_bf16.h>
#include <cstdint>
#include <cstddef>

// ---------------------------------------------------------------------------
// CDNA5 / gfx1250 implementation of GraphSAGE(2) + BiLSTM(2) + bmm scorer.
// Dense algebra: v_wmma_f32_16x16x32_bf16 (f32 accumulate).
// LSTM weight staging: global_load_async_to_lds_b128 + s_wait_asynccnt.
// ---------------------------------------------------------------------------

typedef __bf16 bf16_t;
typedef __attribute__((ext_vector_type(16))) __bf16 v16bf;
typedef __attribute__((ext_vector_type(8)))  __bf16 v8bf;
typedef __attribute__((ext_vector_type(8)))  float  v8f;

#define HN     50000   // graph nodes
#define TT     512     // trace length
#define BB     4       // trace batch
#define LH     128     // per-direction LSTM hidden
#define GATES  512     // 4*LH
#define HID    256     // bilstm output width (2*LH) == sage output width

// ---------------------------------------------------------------------------
// WMMA GEMM:  C[M,N] = A[M,K] * B[N,K]^T (+bias) (+relu)
// A, B row-major bf16; C written as f32 (Cf) and/or bf16 (Cb).
// One wave computes a 16(M) x NSUB*16(N) tile; NSUB is compile-time so the
// inner loop is branch-free (EXEC stays all-ones for WMMA).  All control flow
// is scalar via readfirstlane.  K loop is unrolled x2 into a ping-pong so the
// two fragment buffers alternate structurally (no rotating v_mov copies, no
// WMMA->VALU WAR hazard nops).  Requires K % 64 == 0 (true for all calls:
// K in {128, 256, 512}).
// ---------------------------------------------------------------------------
template <int NSUB>
__global__ __launch_bounds__(256)
void wmma_gemm_kernel(const bf16_t* __restrict__ A,
                      const bf16_t* __restrict__ B,
                      const float*  __restrict__ bias,
                      float*        __restrict__ Cf,
                      bf16_t*       __restrict__ Cb,
                      int M, int N, int K, int relu,
                      int ncol_base, int ntiles)
{
    const int lane = threadIdx.x & 31;
    const int wv   = __builtin_amdgcn_readfirstlane(threadIdx.x >> 5);
    const int wave = blockIdx.x * 8 + wv;          // 8 waves per 256-thr block
    const int mt   = wave / ntiles;
    const int g    = wave - mt * ntiles;
    if (mt * 16 >= M) return;                      // scalar branch

    const int hi   = lane >> 4;                    // lane half
    const int lcol = lane & 15;
    const int row  = mt * 16 + lcol;               // A row for this lane
    const int kgA  = 16 * hi;                      // A lane K-group base
    const int kgB  = 16 * hi;                      // B lane K-group base

    const bf16_t* arow = A + (size_t)row * K;
    const bf16_t* brow[NSUB];
    #pragma unroll
    for (int nn = 0; nn < NSUB; ++nn) {
        const int ncol = ncol_base + (g * NSUB + nn) * 16;
        brow[nn] = B + (size_t)(ncol + lcol) * K + kgB;
    }

    v8f acc[NSUB] = {};

    // A fragment: lane holds row, K = {kg..kg+15}
    auto ldA = [&](int k0) {
        v8bf lo  = *(const v8bf*)(arow + k0 + kgA);
        v8bf hiv = *(const v8bf*)(arow + k0 + kgA + 8);
        v16bf a;
        #pragma unroll
        for (int i = 0; i < 8; ++i) { a[i] = lo[i]; a[i + 8] = hiv[i]; }
        return a;
    };

    // ping buffer (k0 + 0)
    v16bf a0 = ldA(0);
    v16bf b0[NSUB];
    #pragma unroll
    for (int nn = 0; nn < NSUB; ++nn) b0[nn] = *(const v16bf*)(brow[nn]);

    for (int k0 = 0; k0 < K; k0 += 64) {
        // pong buffer loads (k0 + 32) overlap ping WMMAs
        v16bf a1 = ldA(k0 + 32);
        v16bf b1[NSUB];
        #pragma unroll
        for (int nn = 0; nn < NSUB; ++nn)
            b1[nn] = *(const v16bf*)(brow[nn] + k0 + 32);

        #pragma unroll
        for (int nn = 0; nn < NSUB; ++nn)
            acc[nn] = __builtin_amdgcn_wmma_f32_16x16x32_bf16(
                false, a0, false, b0[nn], (short)0, acc[nn], false, false);

        if (k0 + 64 < K) {                         // scalar branch
            // ping buffer reloads (k0 + 64) overlap pong WMMAs
            a0 = ldA(k0 + 64);
            #pragma unroll
            for (int nn = 0; nn < NSUB; ++nn)
                b0[nn] = *(const v16bf*)(brow[nn] + k0 + 64);
        }

        #pragma unroll
        for (int nn = 0; nn < NSUB; ++nn)
            acc[nn] = __builtin_amdgcn_wmma_f32_16x16x32_bf16(
                false, a1, false, b1[nn], (short)0, acc[nn], false, false);
    }

    const int rbase = mt * 16 + 8 * hi;            // C row base for lane half
    #pragma unroll
    for (int nn = 0; nn < NSUB; ++nn) {
        const int ccol = ncol_base + (g * NSUB + nn) * 16 + lcol;
        const float bv = bias ? bias[ccol] : 0.0f;
        #pragma unroll
        for (int r = 0; r < 8; ++r) {
            float v = acc[nn][r] + bv;
            if (relu) v = fmaxf(v, 0.0f);
            const size_t off = (size_t)(rbase + r) * N + ccol;
            if (Cf) Cf[off] = v;
            if (Cb) Cb[off] = (bf16_t)v;
        }
    }
}

// ---------------------------------------------------------------------------
// Graph kernels
// ---------------------------------------------------------------------------
__global__ void degree_kernel(const long long* __restrict__ dst, int E,
                              float* __restrict__ deg)
{
    int e = blockIdx.x * blockDim.x + threadIdx.x;
    if (e < E) atomicAdd(&deg[dst[e]], 1.0f);
}

// one thread per (edge, 4 features): float4 gather + 4 float atomics
__global__ void scatter_kernel(const float* __restrict__ x,
                               const long long* __restrict__ src,
                               const long long* __restrict__ dst,
                               float* __restrict__ agg, int E, int C)
{
    long long tid  = (long long)blockIdx.x * blockDim.x + threadIdx.x;
    int vecs = C >> 2;
    if (tid >= (long long)E * vecs) return;
    int e = (int)(tid / vecs);
    int v = (int)(tid % vecs);
    long long s = src[e], d = dst[e];
    float4 val = ((const float4*)(x + s * (long long)C))[v];
    float* o = agg + d * (long long)C + v * 4;
    atomicAdd(o + 0, val.x);
    atomicAdd(o + 1, val.y);
    atomicAdd(o + 2, val.z);
    atomicAdd(o + 3, val.w);
}

// acat[n] = bf16[ agg[n]/max(deg,1) | self[n] ]   (row width 2C)
__global__ void pack_sage_kernel(const float* __restrict__ agg,
                                 const float* __restrict__ self,
                                 const float* __restrict__ deg,
                                 bf16_t* __restrict__ acat, int Nn, int C)
{
    long long i = (long long)blockIdx.x * blockDim.x + threadIdx.x;
    if (i >= (long long)Nn * C) return;
    int c = (int)(i % C);
    long long n = i / C;
    float dv = fmaxf(deg[n], 1.0f);
    acat[n * 2 * C + c]     = (bf16_t)(agg[n * C + c] / dv);
    acat[n * 2 * C + C + c] = (bf16_t)self[n * C + c];
}

// wcat[o] = bf16[ wl[o] | wr[o] ]  (row width 2C)
__global__ void concat_w_kernel(const float* __restrict__ wl,
                                const float* __restrict__ wr,
                                bf16_t* __restrict__ wcat, int O, int C)
{
    int i = blockIdx.x * blockDim.x + threadIdx.x;
    if (i >= O * 2 * C) return;
    int c = i % (2 * C), o = i / (2 * C);
    float v = (c < C) ? wl[o * C + c] : wr[o * C + (c - C)];
    wcat[(size_t)o * 2 * C + c] = (bf16_t)v;
}

__global__ void f2bf_kernel(const float* __restrict__ in,
                            bf16_t* __restrict__ out, long long n)
{
    long long i = (long long)blockIdx.x * blockDim.x + threadIdx.x;
    if (i < n) out[i] = (bf16_t)in[i];
}

// ---------------------------------------------------------------------------
// Layout shuffles (f32 -> bf16), with optional time reversal.
//  in : [B,T,D] (in_bt=1) or [T,B,D] (in_bt=0)
//  out: rows t*B+b (out_bt=0) or b*T+t (out_bt=1)
// ---------------------------------------------------------------------------
__global__ void repack_kernel(const float* __restrict__ in,
                              bf16_t* __restrict__ out,
                              int B, int T, int D,
                              int in_bt, int reverse, int out_bt)
{
    long long i = (long long)blockIdx.x * blockDim.x + threadIdx.x;
    long long tot = (long long)B * T * D;
    if (i >= tot) return;
    int d = (int)(i % D);
    long long r = i / D;
    int b = (int)(r % B);
    int t = (int)(r / B);
    int tin = reverse ? (T - 1 - t) : t;
    long long io = in_bt  ? ((long long)b * T + tin) * D + d
                          : ((long long)tin * B + b) * D + d;
    long long oo = out_bt ? ((long long)b * T + t) * D + d
                          : ((long long)t * B + b) * D + d;
    out[oo] = (bf16_t)in[io];
}

// ---------------------------------------------------------------------------
// Sequential LSTM recurrence. One workgroup (512 threads) per (layer, dir).
// w_hh [512,128] f32 staged into LDS (256 KB of the 320 KB WGP pool) with
// CDNA5 async-to-LDS DMA (ASYNCcnt-tracked), then consumed 512 steps.
// pre: [T*B, 512] rows t*B+b (already x@w_ih^T + b).
// out: [T,B,256] f32, this direction writing columns [out_off, out_off+128).
// ---------------------------------------------------------------------------
__device__ __forceinline__ float sigf(float x) { return 1.0f / (1.0f + __expf(-x)); }

__global__ __launch_bounds__(512)
void lstm_seq_kernel(const float* __restrict__ pre,
                     const float* __restrict__ whh,
                     float* __restrict__ out,
                     int out_off, int reverse)
{
    extern __shared__ float lds[];
    float* s_w = lds;                     // GATES*LH
    float* s_h = s_w + GATES * LH;        // BB*LH (=512)
    float* s_c = s_h + BB * LH;           // BB*LH
    float* s_g = s_c + BB * LH;           // BB*GATES

    const int tid = threadIdx.x;          // == gate index, and == b*LH+h

    // Stage w_hh (64K floats) into LDS: 16B per lane per issue via the CDNA5
    // async DMA path (no VGPR round-trip); completion via ASYNCcnt.
    for (int i = tid * 4; i < GATES * LH; i += 512 * 4) {
        unsigned           ldsa = (unsigned)(uintptr_t)(s_w + i);
        unsigned long long ga   = (unsigned long long)(uintptr_t)(whh + i);
        asm volatile("global_load_async_to_lds_b128 %0, %1, off"
                     :: "v"(ldsa), "v"(ga) : "memory");
    }
    asm volatile("s_wait_asynccnt 0x0" ::: "memory");

    s_h[tid] = 0.0f;
    s_c[tid] = 0.0f;
    __syncthreads();

    const float* wrow = s_w + tid * LH;
    for (int t = 0; t < TT; ++t) {
        // gate pre-activations: g[b][tid] = pre + h[b] . whh[tid]
        #pragma unroll
        for (int b = 0; b < BB; ++b) {
            float acc = pre[((size_t)(t * BB + b)) * GATES + tid];
            const float* hb = s_h + b * LH;
            #pragma unroll 8
            for (int k = 0; k < LH; ++k) acc = fmaf(wrow[k], hb[k], acc);
            s_g[b * GATES + tid] = acc;
        }
        __syncthreads();
        // cell/hidden update: thread tid owns (b = tid>>7, h = tid&127)
        {
            const int b = tid >> 7, h = tid & (LH - 1);
            const float* g = s_g + b * GATES;
            float iv = g[h], fv = g[LH + h], gv = g[2 * LH + h], ov = g[3 * LH + h];
            float c  = sigf(fv) * s_c[tid] + sigf(iv) * tanhf(gv);
            float hn = sigf(ov) * tanhf(c);
            s_c[tid] = c;
            s_h[tid] = hn;
            int to = reverse ? (TT - 1 - t) : t;
            out[((size_t)(to * BB + b)) * HID + out_off + h] = hn;
        }
        __syncthreads();
    }
}

// ---------------------------------------------------------------------------
// Host side
// ---------------------------------------------------------------------------
static inline void launch_gemm(const bf16_t* A, const bf16_t* B, const float* bias,
                               float* Cf, bf16_t* Cb, int M, int N, int K, int relu,
                               hipStream_t s)
{
    const int Mt  = M / 16;           // M always a multiple of 16 here
    const int nt4 = N / 64;           // full 16x64 tiles
    if (nt4 > 0) {
        long long waves = (long long)Mt * nt4;
        unsigned blocks = (unsigned)((waves + 7) / 8);
        wmma_gemm_kernel<4><<<blocks, 256, 0, s>>>(A, B, bias, Cf, Cb,
                                                   M, N, K, relu, 0, nt4);
    }
    const int rem = (N - nt4 * 64) / 16;  // leftover 16-wide subtiles (0..3)
    if (rem > 0) {
        unsigned blocks = (unsigned)(((long long)Mt + 7) / 8);
        const int base = nt4 * 64;
        switch (rem) {
        case 1: wmma_gemm_kernel<1><<<blocks, 256, 0, s>>>(A, B, bias, Cf, Cb,
                                                           M, N, K, relu, base, 1); break;
        case 2: wmma_gemm_kernel<2><<<blocks, 256, 0, s>>>(A, B, bias, Cf, Cb,
                                                           M, N, K, relu, base, 1); break;
        default: wmma_gemm_kernel<3><<<blocks, 256, 0, s>>>(A, B, bias, Cf, Cb,
                                                            M, N, K, relu, base, 1); break;
        }
    }
}

extern "C" void kernel_launch(void* const* d_in, const int* in_sizes, int n_in,
                              void* d_out, int out_size, void* d_ws, size_t ws_size,
                              hipStream_t stream)
{
    (void)n_in; (void)out_size; (void)ws_size;

    const float*     trace = (const float*)d_in[0];      // [4,512,128]
    const float*     x     = (const float*)d_in[1];      // [50000,128]
    const long long* ei    = (const long long*)d_in[2];  // [2,800000] int64
    const float* s1_wl = (const float*)d_in[3];
    const float* s1_wr = (const float*)d_in[4];
    const float* s1_b  = (const float*)d_in[5];
    const float* s2_wl = (const float*)d_in[6];
    const float* s2_wr = (const float*)d_in[7];
    const float* s2_b  = (const float*)d_in[8];
    const float* wih0f = (const float*)d_in[9];
    const float* whh0f = (const float*)d_in[10];
    const float* b0f   = (const float*)d_in[11];
    const float* wih0r = (const float*)d_in[12];
    const float* whh0r = (const float*)d_in[13];
    const float* b0r   = (const float*)d_in[14];
    const float* wih1f = (const float*)d_in[15];
    const float* whh1f = (const float*)d_in[16];
    const float* b1f   = (const float*)d_in[17];
    const float* wih1r = (const float*)d_in[18];
    const float* whh1r = (const float*)d_in[19];
    const float* b1r   = (const float*)d_in[20];

    const int E = in_sizes[2] / 2;
    const long long* src = ei;
    const long long* dst = ei + E;

    // ---- workspace carve-up (256B aligned) --------------------------------
    char* ws = (char*)d_ws;
    size_t off = 0;
    auto wsalloc = [&](size_t bytes) -> char* {
        char* p = ws + off;
        off += (bytes + 255) & ~(size_t)255;
        return p;
    };
    float*  deg   = (float*) wsalloc((size_t)HN * 4);
    float*  agg   = (float*) wsalloc((size_t)HN * 256 * 4);  // reused L1/L2
    bf16_t* acat  = (bf16_t*)wsalloc((size_t)HN * 512 * 2);  // reused L1/L2
    float*  h1    = (float*) wsalloc((size_t)HN * 256 * 4);
    bf16_t* emb   = (bf16_t*)wsalloc((size_t)HN * 256 * 2);
    bf16_t* w1c   = (bf16_t*)wsalloc((size_t)256 * 256 * 2);
    bf16_t* w2c   = (bf16_t*)wsalloc((size_t)256 * 512 * 2);
    bf16_t* bih0f = (bf16_t*)wsalloc((size_t)GATES * 128 * 2);
    bf16_t* bih0r = (bf16_t*)wsalloc((size_t)GATES * 128 * 2);
    bf16_t* bih1f = (bf16_t*)wsalloc((size_t)GATES * 256 * 2);
    bf16_t* bih1r = (bf16_t*)wsalloc((size_t)GATES * 256 * 2);
    bf16_t* xsb   = (bf16_t*)wsalloc((size_t)TT * BB * 256 * 2); // reused
    float*  pre_f = (float*) wsalloc((size_t)TT * BB * GATES * 4);
    float*  pre_r = (float*) wsalloc((size_t)TT * BB * GATES * 4);
    float*  enc0  = (float*) wsalloc((size_t)TT * BB * HID * 4);
    float*  enc1  = (float*) wsalloc((size_t)TT * BB * HID * 4);
    bf16_t* encb  = (bf16_t*)wsalloc((size_t)TT * BB * HID * 2);

    const int SHM_LSTM = (GATES * LH + 2 * BB * LH + BB * GATES) * 4; // 274,432 B
    hipFuncSetAttribute(reinterpret_cast<const void*>(lstm_seq_kernel),
                        hipFuncAttributeMaxDynamicSharedMemorySize, SHM_LSTM);

    const int TPB = 256;
    auto nb = [](long long n, int tpb) { return (unsigned)((n + tpb - 1) / tpb); };

    // ---- weight conversions (bf16) ---------------------------------------
    concat_w_kernel<<<nb(256 * 256, TPB), TPB, 0, stream>>>(s1_wl, s1_wr, w1c, 256, 128);
    concat_w_kernel<<<nb(256 * 512, TPB), TPB, 0, stream>>>(s2_wl, s2_wr, w2c, 256, 256);
    f2bf_kernel<<<nb(GATES * 128, TPB), TPB, 0, stream>>>(wih0f, bih0f, GATES * 128);
    f2bf_kernel<<<nb(GATES * 128, TPB), TPB, 0, stream>>>(wih0r, bih0r, GATES * 128);
    f2bf_kernel<<<nb(GATES * 256, TPB), TPB, 0, stream>>>(wih1f, bih1f, GATES * 256);
    f2bf_kernel<<<nb(GATES * 256, TPB), TPB, 0, stream>>>(wih1r, bih1r, GATES * 256);

    // ---- GraphSAGE layer 1 -----------------------------------------------
    hipMemsetAsync(deg, 0, (size_t)HN * 4, stream);
    hipMemsetAsync(agg, 0, (size_t)HN * 128 * 4, stream);
    degree_kernel<<<nb(E, TPB), TPB, 0, stream>>>(dst, E, deg);
    scatter_kernel<<<nb((long long)E * 32, TPB), TPB, 0, stream>>>(x, src, dst, agg, E, 128);
    pack_sage_kernel<<<nb((long long)HN * 128, TPB), TPB, 0, stream>>>(agg, x, deg, acat, HN, 128);
    launch_gemm(acat, w1c, s1_b, h1, nullptr, HN, 256, 256, /*relu*/1, stream);

    // ---- GraphSAGE layer 2 -----------------------------------------------
    hipMemsetAsync(agg, 0, (size_t)HN * 256 * 4, stream);
    scatter_kernel<<<nb((long long)E * 64, TPB), TPB, 0, stream>>>(h1, src, dst, agg, E, 256);
    pack_sage_kernel<<<nb((long long)HN * 256, TPB), TPB, 0, stream>>>(agg, h1, deg, acat, HN, 256);
    launch_gemm(acat, w2c, s2_b, nullptr, emb, HN, 256, 512, /*relu*/1, stream);

    // ---- BiLSTM layer 0 (input trace [B,T,128]) --------------------------
    const long long TBD0 = (long long)TT * BB * 128;
    repack_kernel<<<nb(TBD0, TPB), TPB, 0, stream>>>(trace, xsb, BB, TT, 128, 1, 0, 0);
    launch_gemm(xsb, bih0f, b0f, pre_f, nullptr, TT * BB, GATES, 128, 0, stream);
    lstm_seq_kernel<<<1, 512, SHM_LSTM, stream>>>(pre_f, whh0f, enc0, 0, 0);

    repack_kernel<<<nb(TBD0, TPB), TPB, 0, stream>>>(trace, xsb, BB, TT, 128, 1, 1, 0);
    launch_gemm(xsb, bih0r, b0r, pre_r, nullptr, TT * BB, GATES, 128, 0, stream);
    lstm_seq_kernel<<<1, 512, SHM_LSTM, stream>>>(pre_r, whh0r, enc0, LH, 1);

    // ---- BiLSTM layer 1 (input enc0 [T,B,256]) ---------------------------
    const long long TBD1 = (long long)TT * BB * HID;
    repack_kernel<<<nb(TBD1, TPB), TPB, 0, stream>>>(enc0, xsb, BB, TT, HID, 0, 0, 0);
    launch_gemm(xsb, bih1f, b1f, pre_f, nullptr, TT * BB, GATES, HID, 0, stream);
    lstm_seq_kernel<<<1, 512, SHM_LSTM, stream>>>(pre_f, whh1f, enc1, 0, 0);

    repack_kernel<<<nb(TBD1, TPB), TPB, 0, stream>>>(enc0, xsb, BB, TT, HID, 0, 1, 0);
    launch_gemm(xsb, bih1r, b1r, pre_r, nullptr, TT * BB, GATES, HID, 0, stream);
    lstm_seq_kernel<<<1, 512, SHM_LSTM, stream>>>(pre_r, whh1r, enc1, LH, 1);

    // ---- final scorer: out[b,t,n] = enc1[t,b,:] . emb[n,:] ---------------
    repack_kernel<<<nb(TBD1, TPB), TPB, 0, stream>>>(enc1, encb, BB, TT, HID, 0, 0, 1);
    launch_gemm(encb, emb, nullptr, (float*)d_out, nullptr, BB * TT, HN, HID, 0, stream);
}
